// BinaryCNN_75514114998340
// MI455X (gfx1250) — matrix operand order repaired and checked by reading, not compiled, not run on hardware
//
#include <hip/hip_runtime.h>
#include <hip/hip_bf16.h>
#include <math.h>

typedef __attribute__((ext_vector_type(16))) _Float16 v16h;
typedef __attribute__((ext_vector_type(8)))  _Float16 v8h;
typedef __attribute__((ext_vector_type(8)))  float    v8f;

#define EPSBN 1e-5f

// ---------------------------------------------------------------------------
// Zero fp16 buffer (8 halves/thread, grid exact)
// ---------------------------------------------------------------------------
__global__ void zero_kernel(_Float16* __restrict__ p) {
    size_t t = (size_t)blockIdx.x * blockDim.x + threadIdx.x;
    *(v8h*)(p + t * 8) = (v8h){};
}

// ---------------------------------------------------------------------------
// Per-output-channel quant scale: s = max(absmax/127, 1e-8)
// ---------------------------------------------------------------------------
__global__ void rowmax_kernel(const float* __restrict__ w, float* __restrict__ scales, int K) {
    __shared__ float red[256];
    int n = blockIdx.x;
    const float* row = w + (size_t)n * K;
    float m = 0.f;
    for (int k = threadIdx.x; k < K; k += 256) m = fmaxf(m, fabsf(row[k]));
    red[threadIdx.x] = m;
    __syncthreads();
    for (int s = 128; s > 0; s >>= 1) {
        if (threadIdx.x < s) red[threadIdx.x] = fmaxf(red[threadIdx.x], red[threadIdx.x + s]);
        __syncthreads();
    }
    if (threadIdx.x == 0) scales[n] = fmaxf(red[0] * (1.f / 127.f), 1e-8f);
}

// ---------------------------------------------------------------------------
// Fold conv bias + BatchNorm into per-channel affine: y = x*sc + bi
// ---------------------------------------------------------------------------
__global__ void affine_kernel(const float* __restrict__ b, const float* __restrict__ g,
                              const float* __restrict__ be, const float* __restrict__ mu,
                              const float* __restrict__ va,
                              float* __restrict__ sc, float* __restrict__ bi,
                              int N, int Npad, int has_bn) {
    int n = blockIdx.x * blockDim.x + threadIdx.x;
    if (n >= Npad) return;
    if (n >= N) { sc[n] = 0.f; bi[n] = 0.f; return; }
    if (has_bn) {
        float s = g[n] * rsqrtf(va[n] + EPSBN);
        sc[n] = s;
        bi[n] = (b[n] - mu[n]) * s + be[n];
    } else {
        sc[n] = 1.f;
        bi[n] = b[n];
    }
}

// ---------------------------------------------------------------------------
// Quant-dequant weights into WMMA B-fragment-major layout (fp16), with a
// K-permutation so GEMM K-order is (kh,kw,ci) [ci innermost, NHWC]:
//   k_mem = s*inner + ci  ->  k_ref = (k_mem % inner)*outer + k_mem/inner
// B 32x16 fragment order used here (A matches): lanes 0-15 hold K=0..15, lanes 16-31 K=16..31.
// ---------------------------------------------------------------------------
__global__ void pack_kernel(const float* __restrict__ w, const float* __restrict__ scales,
                            _Float16* __restrict__ packed, int N, int K, int KC,
                            int inner, int outer) {
    int tid  = blockIdx.x * blockDim.x + threadIdx.x;
    int e    = tid & 15;
    int lane = (tid >> 4) & 31;
    int rest = tid >> 9;
    int kc   = rest % KC;
    int nt   = rest / KC;
    int n = nt * 16 + (lane & 15);
    int k = kc * 32 + (lane >> 4) * 16 + e;        // memory-order K index
    int kout = k / inner;
    int kin  = k - kout * inner;
    int kref = kin * outer + kout;                 // reference (OIHW / flatten) order
    float v = 0.f;
    if (n < N) {
        float s = scales[n];
        float q = rintf(w[(size_t)n * K + kref] / s);   // RNE, matches jnp.round
        q = fminf(fmaxf(q, -128.f), 127.f);
        v = q * s;
    }
    packed[tid] = (_Float16)v;
}

// ---------------------------------------------------------------------------
// conv1: 1->32, 5x5, pad 2, fp32 direct (tiny), fused BN+hardtanh.
// Input [B,1,28,28]; output fp16 NHWC with halo pad 2: [B,32,32,32]
// (halo pre-zeroed by zero_kernel).
// ---------------------------------------------------------------------------
__global__ void conv1_kernel(const float* __restrict__ x, const float* __restrict__ w,
                             const float* __restrict__ scales,
                             const float* __restrict__ sc, const float* __restrict__ bi,
                             _Float16* __restrict__ out) {
    int t = blockIdx.x * blockDim.x + threadIdx.x;   // 1024*28*28*32 interior
    int c  = t & 31;
    int r  = t >> 5;
    int ow = r % 28; r /= 28;
    int oh = r % 28;
    int b  = r / 28;
    float s = scales[c];
    const float* xb = x + (size_t)b * 784;
    const float* wc = w + c * 25;
    float acc = 0.f;
#pragma unroll
    for (int kh = 0; kh < 5; ++kh) {
        int ih = oh - 2 + kh;
        if ((unsigned)ih >= 28u) continue;
#pragma unroll
        for (int kw = 0; kw < 5; ++kw) {
            int iw = ow - 2 + kw;
            if ((unsigned)iw >= 28u) continue;
            float q = rintf(wc[kh * 5 + kw] / s);
            q = fminf(fmaxf(q, -128.f), 127.f) * s;
            acc += xb[ih * 28 + iw] * q;
        }
    }
    float y = acc * sc[c] + bi[c];
    y = fminf(fmaxf(y, -1.f), 1.f);
    out[(((size_t)b * 32 + oh + 2) * 32 + (ow + 2)) * 32 + c] = (_Float16)y;
}

// ---------------------------------------------------------------------------
// Implicit-GEMM conv / FC via v_wmma_f32_16x16x32_f16.
// NHWC activations with zero halo [B, H+2P, W+2P, CIN] -> all A-loads are
// unconditional 16B b128s (no bounds checks / exec masking). GEMM K-order
// (kh,kw,ci). Each wave owns a 2Mx2N register tile: per K-step 8 b128 loads
// feed 4 WMMAs. Output: unpadded NHWC fp16 (or fp32 [B,Nvalid] for fc3).
// ---------------------------------------------------------------------------
template<int CIN, int H, int W, int KH, int KW, int PAD>
__global__ __launch_bounds__(128) void wmma_conv_nhwc(
    const _Float16* __restrict__ in, const _Float16* __restrict__ wp,
    const float* __restrict__ sc, const float* __restrict__ bi,
    _Float16* __restrict__ out_h, float* __restrict__ out_f,
    int Cout, int KC, int apply_ht, int Nvalid) {
    constexpr int HW = H * W;
    constexpr int Hp = H + 2 * PAD;
    constexpr int Wp = W + 2 * PAD;
    const int lane = threadIdx.x & 31;
    const int lm   = lane & 15;
    const int kgrp = lane >> 4;
    const int mt0  = (blockIdx.x * 4 + (threadIdx.x >> 5)) * 2;  // 2 consecutive M tiles
    const int nt0  = blockIdx.y * 2;                             // 2 consecutive N tiles

    int ohI[2], owI[2], bI[2];
    const _Float16* inb[2];
#pragma unroll
    for (int i = 0; i < 2; ++i) {
        int m  = (mt0 + i) * 16 + lm;
        int b  = m / HW;                 // constant divisor
        int mr = m - b * HW;
        ohI[i] = mr / W;
        owI[i] = mr - ohI[i] * W;
        bI[i]  = b;
        inb[i] = in + (size_t)b * Hp * Wp * CIN;
    }

    v8f acc[2][2] = {};
    for (int kc = 0; kc < KC; ++kc) {
        // two B fragments (32B each) from prepacked weights
        v16h bf[2];
#pragma unroll
        for (int j = 0; j < 2; ++j)
            bf[j] = *(const v16h*)(wp + ((size_t)((nt0 + j) * KC + kc) * 32 + lane) * 16);

        // two A fragments: 2 runs x 2 m-tiles of unconditional 16B loads
        const int kbase = kc * 32 + kgrp * 16;
        v8h run[2][2];                   // [mtile][run]
#pragma unroll
        for (int rn = 0; rn < 2; ++rn) {
            int k0 = kbase + rn * 8;
            int s  = k0 / CIN;           // constant divisor (pow2 for convs)
            int cb = k0 - s * CIN;       // 8-aligned channel base
            int kh = s / KW;             // constant divisor
            int kw = s - kh * KW;
#pragma unroll
            for (int i = 0; i < 2; ++i) {
                // halo-padded plane: ih = ohI+kh in [0, Hp), always in bounds
                run[i][rn] = *(const v8h*)(inb[i] +
                    ((size_t)(ohI[i] + kh) * Wp + (owI[i] + kw)) * CIN + cb);
            }
        }
        v16h af[2];
#pragma unroll
        for (int i = 0; i < 2; ++i)
            af[i] = __builtin_shufflevector(run[i][0], run[i][1],
                        0,1,2,3,4,5,6,7,8,9,10,11,12,13,14,15);
#pragma unroll
        for (int i = 0; i < 2; ++i)
#pragma unroll
            for (int j = 0; j < 2; ++j)
                acc[i][j] = __builtin_amdgcn_wmma_f32_16x16x32_f16(
                    false, af[i], false, bf[j], (short)0, acc[i][j], false, false);
    }

#pragma unroll
    for (int j = 0; j < 2; ++j) {
        const int n = (nt0 + j) * 16 + lm;
        const float s0 = sc[n];
        const float t0 = bi[n];
#pragma unroll
        for (int i = 0; i < 2; ++i) {
#pragma unroll
            for (int r2 = 0; r2 < 8; ++r2) {
                int mrow = (mt0 + i) * 16 + kgrp * 8 + r2;  // C/D: VGPR r -> M=r+8*kgrp
                float y = acc[i][j][r2] * s0 + t0;
                if (apply_ht) y = fminf(fmaxf(y, -1.f), 1.f);
                int b2 = mrow / HW;
                int rr = mrow - b2 * HW;
                if (out_f) {
                    if (n < Nvalid) out_f[(size_t)b2 * Nvalid + n] = y;
                } else {
                    out_h[((size_t)b2 * HW + rr) * Cout + n] = (_Float16)y;  // NHWC
                }
            }
        }
    }
}

// ---------------------------------------------------------------------------
// 2x2 maxpool, stride 2, VALID, fp16 NHWC; writes into halo-padded output
// plane [B, Hpo, Wpo, C] at interior offset Po (halo pre-zeroed).
// BN/hardtanh already applied; commutes with max since BN scale > 0.
// ---------------------------------------------------------------------------
__global__ void pool_kernel(const _Float16* __restrict__ in, _Float16* __restrict__ out,
                            int C, int H, int W, int OH, int OW,
                            int Po, int Hpo, int Wpo) {
    int t = blockIdx.x * blockDim.x + threadIdx.x;   // B*OH*OW*C (grid exact)
    int c  = t % C;  int r = t / C;
    int ox = r % OW; r /= OW;
    int oy = r % OH; int b = r / OH;
    const _Float16* p = in + (((size_t)b * H + oy * 2) * W + ox * 2) * C + c;
    float a0 = (float)p[0],             a1 = (float)p[C];
    float a2 = (float)p[(size_t)W * C], a3 = (float)p[(size_t)W * C + C];
    out[(((size_t)b * Hpo + oy + Po) * Wpo + ox + Po) * C + c] =
        (_Float16)fmaxf(fmaxf(a0, a1), fmaxf(a2, a3));
}

// ---------------------------------------------------------------------------
extern "C" void kernel_launch(void* const* d_in, const int* in_sizes, int n_in,
                              void* d_out, int out_size, void* d_ws, size_t ws_size,
                              hipStream_t stream) {
    (void)in_sizes; (void)n_in; (void)out_size; (void)ws_size;
    const float* x = (const float*)d_in[0];

    const float *w[8], *bb[8], *g[8], *be[8], *mu[8], *va[8];
    for (int i = 0; i < 5; ++i) {           // conv1..conv5: indices 1..30
        int base = 1 + 6 * i;
        w[i]  = (const float*)d_in[base + 0];
        bb[i] = (const float*)d_in[base + 1];
        g[i]  = (const float*)d_in[base + 2];
        be[i] = (const float*)d_in[base + 3];
        mu[i] = (const float*)d_in[base + 4];
        va[i] = (const float*)d_in[base + 5];
    }
    for (int i = 0; i < 2; ++i) {           // fc1, fc2: indices 31..42
        int base = 31 + 6 * i;
        w[5 + i]  = (const float*)d_in[base + 0];
        bb[5 + i] = (const float*)d_in[base + 1];
        g[5 + i]  = (const float*)d_in[base + 2];
        be[5 + i] = (const float*)d_in[base + 3];
        mu[5 + i] = (const float*)d_in[base + 4];
        va[5 + i] = (const float*)d_in[base + 5];
    }
    w[7] = (const float*)d_in[43];  bb[7] = (const float*)d_in[44];   // fc3 (no BN)
    g[7] = be[7] = mu[7] = va[7] = bb[7];                             // unused

    static const int N_[8]    = {32, 64, 128, 256, 256, 1024, 512, 10};
    static const int Npad_[8] = {32, 64, 128, 256, 256, 1024, 512, 32};  // fc3 -> 2 N-tiles
    static const int K_[8]    = {25, 800, 576, 1152, 256, 2304, 1024, 512};
    static const int Pin_[8]  = {25, 32, 64, 128, 256, 256, 1024, 512};
    static const int Pout_[8] = {1, 25, 9, 9, 1, 9, 1, 1};

    // ---- workspace layout ----
    char* ws = (char*)d_ws;
    size_t off = 0;
    auto bump = [&](size_t bytes) -> size_t {
        size_t o = off; off += (bytes + 255) & ~(size_t)255; return o;
    };
    float *scales[8], *scv[8], *biv[8];
    _Float16* wp[8];
    for (int l = 0; l < 8; ++l) {
        scales[l] = (float*)(ws + bump((size_t)N_[l] * 4));
        scv[l]    = (float*)(ws + bump((size_t)Npad_[l] * 4));
        biv[l]    = (float*)(ws + bump((size_t)Npad_[l] * 4));
    }
    wp[0] = nullptr;
    for (int l = 1; l < 8; ++l)
        wp[l] = (_Float16*)(ws + bump((size_t)Npad_[l] * K_[l] * 2));   // ~6.7 MB
    const size_t MB = 1024ull * 1024ull;
    size_t actoff = (off + 255) & ~(size_t)255;
    _Float16* actA = (_Float16*)(ws + actoff);             // <= 64 MiB region
    _Float16* actB = (_Float16*)(ws + actoff + 68 * MB);   // <= 103 MB region

    // ---- weight preprocessing ----
    for (int l = 0; l < 8; ++l) {
        rowmax_kernel<<<N_[l], 256, 0, stream>>>(w[l], scales[l], K_[l]);
        affine_kernel<<<(Npad_[l] + 255) / 256, 256, 0, stream>>>(
            bb[l], g[l], be[l], mu[l], va[l], scv[l], biv[l], N_[l], Npad_[l], l < 7 ? 1 : 0);
        if (l >= 1) {
            int total = Npad_[l] * K_[l];   // divisible by 256
            pack_kernel<<<total / 256, 256, 0, stream>>>(
                w[l], scales[l], wp[l], N_[l], K_[l], K_[l] / 32, Pin_[l], Pout_[l]);
        }
    }

    // ---- conv1: h1p [1024,32,32,32] padded NHWC in actA (halo zeroed) ----
    zero_kernel<<<(1024 * 32 * 32 * 32) / (8 * 256), 256, 0, stream>>>(actA);
    conv1_kernel<<<(1024 * 28 * 28 * 32) / 256, 256, 0, stream>>>(
        x, w[0], scales[0], scv[0], biv[0], actA);

    // ---- conv2: 32->64, 5x5 pad2 -> c2pre [1024,28,28,64] in actB ----
    wmma_conv_nhwc<32, 28, 28, 5, 5, 2><<<dim3(6272, 2), 128, 0, stream>>>(
        actA, wp[1], scv[1], biv[1], actB, nullptr, 64, 800 / 32, 1, 64);
    // h2p [1024,16,16,64] padded in actA
    zero_kernel<<<(1024 * 16 * 16 * 64) / (8 * 256), 256, 0, stream>>>(actA);
    pool_kernel<<<(1024 * 14 * 14 * 64) / 256, 256, 0, stream>>>(
        actB, actA, 64, 28, 28, 14, 14, 1, 16, 16);

    // ---- conv3: 64->128, 3x3 pad1 -> c3pre [1024,14,14,128] in actB ----
    wmma_conv_nhwc<64, 14, 14, 3, 3, 1><<<dim3(1568, 4), 128, 0, stream>>>(
        actA, wp[2], scv[2], biv[2], actB, nullptr, 128, 576 / 32, 1, 128);
    // h3p [1024,9,9,128] padded in actA
    zero_kernel<<<(1024 * 9 * 9 * 128) / (8 * 256), 256, 0, stream>>>(actA);
    pool_kernel<<<(1024 * 7 * 7 * 128) / 256, 256, 0, stream>>>(
        actB, actA, 128, 14, 14, 7, 7, 1, 9, 9);

    // ---- conv4: 128->256, 3x3 pad1 -> c4pre [1024,7,7,256] in actB ----
    wmma_conv_nhwc<128, 7, 7, 3, 3, 1><<<dim3(392, 8), 128, 0, stream>>>(
        actA, wp[3], scv[3], biv[3], actB, nullptr, 256, 1152 / 32, 1, 256);
    // h4 [1024,3,3,256] (no halo) in actA
    pool_kernel<<<(1024 * 3 * 3 * 256) / 256, 256, 0, stream>>>(
        actB, actA, 256, 7, 7, 3, 3, 0, 3, 3);

    // ---- conv5: 256->256, 1x1 -> h5 [1024,3,3,256] in actB ----
    wmma_conv_nhwc<256, 3, 3, 1, 1, 0><<<dim3(72, 8), 128, 0, stream>>>(
        actA, wp[4], scv[4], biv[4], actB, nullptr, 256, 256 / 32, 1, 256);

    // ---- fc1: 2304->1024 (weights K-permuted for NHWC flatten) ----
    wmma_conv_nhwc<2304, 1, 1, 1, 1, 0><<<dim3(8, 32), 128, 0, stream>>>(
        actB, wp[5], scv[5], biv[5], actA, nullptr, 1024, 2304 / 32, 1, 1024);

    // ---- fc2: 1024->512 ----
    wmma_conv_nhwc<1024, 1, 1, 1, 1, 0><<<dim3(8, 16), 128, 0, stream>>>(
        actA, wp[6], scv[6], biv[6], actB, nullptr, 512, 1024 / 32, 1, 512);

    // ---- fc3: 512->10, fp32 output [1024,10], no BN/ht ----
    wmma_conv_nhwc<512, 1, 1, 1, 1, 0><<<dim3(8, 1), 128, 0, stream>>>(
        actB, wp[7], scv[7], biv[7], nullptr, (float*)d_out, 32, 512 / 32, 0, 10);
}
